// MoveAutoEncoder_45535243272625
// MI455X (gfx1250) — matrix operand order repaired and checked by reading, not compiled, not run on hardware
//
#include <hip/hip_runtime.h>

typedef __bf16 bf16_t;
typedef bf16_t v16bf __attribute__((ext_vector_type(16)));
typedef bf16_t v8bf  __attribute__((ext_vector_type(8)));
typedef bf16_t v4bf  __attribute__((ext_vector_type(4)));
typedef float  v8f   __attribute__((ext_vector_type(8)));

#define BATCH  4096
#define DIM_IN 2048
#define DIM_H  1024
#define NE_    1024
#define ED_    8

// ---------------------------------------------------------------------------
// Elementwise f32 -> bf16 conversion (x -> xb), 4 elems/thread
// ---------------------------------------------------------------------------
__global__ __launch_bounds__(256)
void cvt_bf16_kernel(const float* __restrict__ x, bf16_t* __restrict__ out, int n4)
{
    int i = blockIdx.x * 256 + threadIdx.x;
    if (i < n4) {
        float4 v = ((const float4*)x)[i];
        v4bf o = { (bf16_t)v.x, (bf16_t)v.y, (bf16_t)v.z, (bf16_t)v.w };
        *(v4bf*)(out + (size_t)i * 4) = o;
    }
}

// ---------------------------------------------------------------------------
// Transpose + convert: W f32 [K,N] row-major  ->  Wt bf16 [N,K] row-major
// ---------------------------------------------------------------------------
__global__ __launch_bounds__(256)
void transpose_cvt_kernel(const float* __restrict__ W, bf16_t* __restrict__ Wt,
                          int K, int N)
{
    __shared__ float tile[32][33];
    const int tx = threadIdx.x & 31;
    const int ty = threadIdx.x >> 5;          // 0..7
    const int n0 = blockIdx.x * 32;
    const int k0 = blockIdx.y * 32;
#pragma unroll
    for (int i = 0; i < 4; ++i)
        tile[ty + i * 8][tx] = W[(size_t)(k0 + ty + i * 8) * N + (n0 + tx)];
    __syncthreads();
#pragma unroll
    for (int i = 0; i < 4; ++i)
        Wt[(size_t)(n0 + ty + i * 8) * K + (k0 + tx)] = (bf16_t)tile[tx][ty + i * 8];
}

// ---------------------------------------------------------------------------
// Fused GEMM + bias + (LayerNorm + ReLU) epilogue.
//   A  : bf16 [M,K] row-major      Wt : bf16 [N,K] row-major (pre-transposed)
//   block = 256 threads (8 waves); block tile = 32 rows x 1024 cols
//   each wave: 2 M-slabs x 8 N-tiles -> 16x v_wmma_f32_16x16x32_bf16 / K-step,
//   every B fragment feeds two WMMAs (halves the weight-stream bytes/FLOP).
//   K templated -> all tile offsets are immediates off one base pointer/stream.
// MODE 0: out_bf  = relu(LN(A@W + b))          (gridDim.y == 1, N == 1024)
// MODE 1: out_f32 = LN(A@W + b)        (z)     (gridDim.y == 1, N == 1024)
// MODE 2: out_f32 = A@W + b            (x_hat) (N == 2048, gridDim.y == 2)
// ---------------------------------------------------------------------------
template<int K, int MODE>
__global__ __launch_bounds__(256)
void gemm_ln_kernel(const bf16_t* __restrict__ A, const bf16_t* __restrict__ Wt,
                    const float* __restrict__ bias, const float* __restrict__ gamma,
                    const float* __restrict__ beta,
                    bf16_t* __restrict__ out_bf, float* __restrict__ out_f32, int N)
{
    const int lane  = threadIdx.x & 31;
    const int wave  = threadIdx.x >> 5;
    const int m16   = lane & 15;
    const int khA   = (lane >> 4) << 3;    // A frag: K offset 0 / 8 per lane-half
    const int khB   = (lane >> 4) << 3;    // B frag: K offset 0 / 8 per lane-half
    const int mrow0 = blockIdx.x * 32;
    const int nbase = blockIdx.y * 1024 + wave * 128;

    const bf16_t* aptr = A  + (size_t)(mrow0 + m16) * K + khA;
    const bf16_t* bptr = Wt + (size_t)(nbase + m16) * K + khB;

    v8f acc0[8] = {};   // rows mrow0 +  0..15
    v8f acc1[8] = {};   // rows mrow0 + 16..31

#pragma unroll 1
    for (int k = 0; k < K; k += 32) {
        v8bf a0lo = *(const v8bf*)(aptr);
        v8bf a0hi = *(const v8bf*)(aptr + 16);
        v8bf a1lo = *(const v8bf*)(aptr + 16 * K);
        v8bf a1hi = *(const v8bf*)(aptr + 16 * K + 16);
        v16bf a0 = __builtin_shufflevector(a0lo, a0hi,
                       0,1,2,3,4,5,6,7,8,9,10,11,12,13,14,15);
        v16bf a1 = __builtin_shufflevector(a1lo, a1hi,
                       0,1,2,3,4,5,6,7,8,9,10,11,12,13,14,15);
        __builtin_prefetch(aptr + 128, 0, 0);
        __builtin_prefetch(aptr + 16 * K + 128, 0, 0);
#pragma unroll
        for (int t = 0; t < 8; ++t) {
            v8bf blo = *(const v8bf*)(bptr + t * 16 * K);   // imm offset
            v8bf bhi = *(const v8bf*)(bptr + t * 16 * K + 16);
            v16bf b = __builtin_shufflevector(blo, bhi,
                       0,1,2,3,4,5,6,7,8,9,10,11,12,13,14,15);
            __builtin_prefetch(bptr + t * 16 * K + 256, 0, 0);
            acc0[t] = __builtin_amdgcn_wmma_f32_16x16x32_bf16(
                          false, a0, false, b, (short)0, acc0[t], false, false);
            acc1[t] = __builtin_amdgcn_wmma_f32_16x16x32_bf16(
                          false, a1, false, b, (short)0, acc1[t], false, false);
        }
        aptr += 32;
        bptr += 32;
    }

    const int rsel = (lane >> 4) << 3;     // row offset within a 16x16 C tile
    float bv[8], gv[8], bev[8];
#pragma unroll
    for (int t = 0; t < 8; ++t) {
        int col = nbase + t * 16 + m16;
        bv[t] = bias[col];
        if constexpr (MODE != 2) { gv[t] = gamma[col]; bev[t] = beta[col]; }
    }

    if constexpr (MODE == 2) {             // bias only -> f32 (x_hat)
        auto emit = [&](v8f (&acc)[8], int mt) {
#pragma unroll
            for (int v = 0; v < 8; ++v) {
                int grow = mrow0 + mt * 16 + v + rsel;
#pragma unroll
                for (int t = 0; t < 8; ++t) {
                    int col = nbase + t * 16 + m16;
                    out_f32[(size_t)grow * N + col] = acc[t][v] + bv[t];
                }
            }
        };
        emit(acc0, 0);
        emit(acc1, 1);
        return;
    } else {
        // ---- fused LayerNorm over the 1024-wide rows (whole row in block) ----
        __shared__ float s_sum[32];
        __shared__ float s_sq [32];
        if (threadIdx.x < 32) { s_sum[threadIdx.x] = 0.f; s_sq[threadIdx.x] = 0.f; }
        __syncthreads();

        auto stats = [&](v8f (&acc)[8], int mt) {
#pragma unroll
            for (int v = 0; v < 8; ++v) {
                float p = 0.f, p2 = 0.f;
#pragma unroll
                for (int t = 0; t < 8; ++t) {
                    float c = acc[t][v] + bv[t];
                    acc[t][v] = c;
                    p += c; p2 += c * c;
                }
                atomicAdd(&s_sum[mt * 16 + v + rsel], p);
                atomicAdd(&s_sq [mt * 16 + v + rsel], p2);
            }
        };
        stats(acc0, 0);
        stats(acc1, 1);
        __syncthreads();

        auto store = [&](v8f (&acc)[8], int mt) {
#pragma unroll
            for (int v = 0; v < 8; ++v) {
                int r     = mt * 16 + v + rsel;
                int grow  = mrow0 + r;
                float mean = s_sum[r] * (1.f / 1024.f);
                float var  = s_sq [r] * (1.f / 1024.f) - mean * mean;
                float rstd = rsqrtf(var + 1e-5f);
#pragma unroll
                for (int t = 0; t < 8; ++t) {
                    int col = nbase + t * 16 + m16;
                    float y = (acc[t][v] - mean) * rstd * gv[t] + bev[t];
                    if constexpr (MODE == 0) {
                        y = fmaxf(y, 0.f);
                        out_bf[(size_t)grow * 1024 + col] = (bf16_t)y;
                    } else {
                        out_f32[(size_t)grow * 1024 + col] = y;
                    }
                }
            }
        };
        store(acc0, 0);
        store(acc1, 1);
    }
}

// ---------------------------------------------------------------------------
// Vector quantization: per 8-elem group, argmin over 1024 codes (LDS-resident),
// write relu(z_q) bf16 for decoder, accumulate sum((z_q - z)^2) for the loss.
// ---------------------------------------------------------------------------
__global__ __launch_bounds__(256)
void vq_kernel(const float* __restrict__ z, const float* __restrict__ codebook,
               bf16_t* __restrict__ dec_in, float* __restrict__ loss_acc)
{
    __shared__ float cb[NE_ * ED_];    // 32 KB
    __shared__ float cnorm[NE_];       //  4 KB
    __shared__ float red[256];

    const int tid = threadIdx.x;
    for (int i = tid; i < NE_ * ED_ / 4; i += 256)
        ((float4*)cb)[i] = ((const float4*)codebook)[i];
    __syncthreads();
    for (int e = tid; e < NE_; e += 256) {
        float s = 0.f;
#pragma unroll
        for (int i = 0; i < ED_; ++i) { float c = cb[e * ED_ + i]; s += c * c; }
        cnorm[e] = s;
    }
    __syncthreads();

    const size_t row = (size_t)blockIdx.x * 256 + tid;
    float zv[ED_];
    {
        float4 a = ((const float4*)(z + row * ED_))[0];
        float4 b = ((const float4*)(z + row * ED_))[1];
        zv[0]=a.x; zv[1]=a.y; zv[2]=a.z; zv[3]=a.w;
        zv[4]=b.x; zv[5]=b.y; zv[6]=b.z; zv[7]=b.w;
    }
    float best = 3.4e38f; int bi = 0;
    for (int e = 0; e < NE_; ++e) {
        float dot = 0.f;
#pragma unroll
        for (int i = 0; i < ED_; ++i) dot = fmaf(cb[e * ED_ + i], zv[i], dot);
        float score = cnorm[e] - 2.f * dot;   // argmin-equivalent to ||z-c||^2
        if (score < best) { best = score; bi = e; }
    }
    float sq = 0.f;
#pragma unroll
    for (int i = 0; i < ED_; ++i) {
        float c = cb[bi * ED_ + i];
        float d = c - zv[i];
        sq += d * d;
        dec_in[row * ED_ + i] = (bf16_t)fmaxf(c, 0.f);   // relu(z_q)
    }
    red[tid] = sq;
    __syncthreads();
    for (int s = 128; s > 0; s >>= 1) {
        if (tid < s) red[tid] += red[tid + s];
        __syncthreads();
    }
    if (tid == 0) atomicAdd(loss_acc, red[0]);
}

__global__ void init_loss_kernel(float* loss) { *loss = 0.f; }

__global__ void finalize_loss_kernel(const float* __restrict__ loss,
                                     float* __restrict__ out)
{
    out[0] = loss[0] * (1.0f + 0.001f) / (float)((size_t)BATCH * DIM_H);
}

// ---------------------------------------------------------------------------
extern "C" void kernel_launch(void* const* d_in, const int* in_sizes, int n_in,
                              void* d_out, int out_size, void* d_ws, size_t ws_size,
                              hipStream_t stream)
{
    const float* x   = (const float*)d_in[0];
    const float* cbk = (const float*)d_in[1];
    const float* W [6] = { (const float*)d_in[2],  (const float*)d_in[6],
                           (const float*)d_in[10], (const float*)d_in[14],
                           (const float*)d_in[18], (const float*)d_in[22] };
    const float* bb[6] = { (const float*)d_in[3],  (const float*)d_in[7],
                           (const float*)d_in[11], (const float*)d_in[15],
                           (const float*)d_in[19], (const float*)d_in[23] };
    const float* gg[5] = { (const float*)d_in[4],  (const float*)d_in[8],
                           (const float*)d_in[12], (const float*)d_in[16],
                           (const float*)d_in[20] };
    const float* be[5] = { (const float*)d_in[5],  (const float*)d_in[9],
                           (const float*)d_in[13], (const float*)d_in[17],
                           (const float*)d_in[21] };

    float* xhat = (float*)d_out;
    float* loss_out = xhat + (size_t)BATCH * DIM_IN;

    char* ws = (char*)d_ws;
    size_t off = 0;
    auto take = [&](size_t bytes) -> char* {
        char* p = ws + off;
        off = (off + bytes + 255) & ~(size_t)255;
        return p;
    };
    bf16_t* xb     = (bf16_t*)take((size_t)BATCH * DIM_IN * 2);
    bf16_t* Wt0    = (bf16_t*)take((size_t)DIM_H * DIM_IN * 2);
    bf16_t* Wt1    = (bf16_t*)take((size_t)DIM_H * DIM_H * 2);
    bf16_t* Wt2    = (bf16_t*)take((size_t)DIM_H * DIM_H * 2);
    bf16_t* Wt3    = (bf16_t*)take((size_t)DIM_H * DIM_H * 2);
    bf16_t* Wt4    = (bf16_t*)take((size_t)DIM_H * DIM_H * 2);
    bf16_t* Wt5    = (bf16_t*)take((size_t)DIM_IN * DIM_H * 2);
    bf16_t* act0   = (bf16_t*)take((size_t)BATCH * DIM_H * 2);
    bf16_t* act1   = (bf16_t*)take((size_t)BATCH * DIM_H * 2);
    float*  zbuf   = (float*) take((size_t)BATCH * DIM_H * 4);
    bf16_t* dec_in = (bf16_t*)take((size_t)BATCH * DIM_H * 2);
    bf16_t* act3   = (bf16_t*)take((size_t)BATCH * DIM_H * 2);
    bf16_t* act4   = (bf16_t*)take((size_t)BATCH * DIM_H * 2);
    float*  lossp  = (float*) take(256);

    // --- stage 0: precision staging (one pass, L2-resident afterwards) ---
    cvt_bf16_kernel<<<(BATCH * DIM_IN / 4 + 255) / 256, 256, 0, stream>>>(
        x, xb, BATCH * DIM_IN / 4);
    transpose_cvt_kernel<<<dim3(DIM_H / 32, DIM_IN / 32), 256, 0, stream>>>(W[0], Wt0, DIM_IN, DIM_H);
    transpose_cvt_kernel<<<dim3(DIM_H / 32, DIM_H  / 32), 256, 0, stream>>>(W[1], Wt1, DIM_H, DIM_H);
    transpose_cvt_kernel<<<dim3(DIM_H / 32, DIM_H  / 32), 256, 0, stream>>>(W[2], Wt2, DIM_H, DIM_H);
    transpose_cvt_kernel<<<dim3(DIM_H / 32, DIM_H  / 32), 256, 0, stream>>>(W[3], Wt3, DIM_H, DIM_H);
    transpose_cvt_kernel<<<dim3(DIM_H / 32, DIM_H  / 32), 256, 0, stream>>>(W[4], Wt4, DIM_H, DIM_H);
    transpose_cvt_kernel<<<dim3(DIM_IN / 32, DIM_H / 32), 256, 0, stream>>>(W[5], Wt5, DIM_H, DIM_IN);
    init_loss_kernel<<<1, 1, 0, stream>>>(lossp);

    const dim3 ggrid(BATCH / 32, 1);

    // --- encoder ---
    gemm_ln_kernel<DIM_IN, 0><<<ggrid, 256, 0, stream>>>(
        xb,   Wt0, bb[0], gg[0], be[0], act0, nullptr, DIM_H);
    gemm_ln_kernel<DIM_H, 0><<<ggrid, 256, 0, stream>>>(
        act0, Wt1, bb[1], gg[1], be[1], act1, nullptr, DIM_H);
    gemm_ln_kernel<DIM_H, 1><<<ggrid, 256, 0, stream>>>(
        act1, Wt2, bb[2], gg[2], be[2], nullptr, zbuf, DIM_H);

    // --- vector quantize + commitment loss + relu(z_q) ---
    vq_kernel<<<(size_t)BATCH * DIM_H / ED_ / 256, 256, 0, stream>>>(
        zbuf, cbk, dec_in, lossp);

    // --- decoder ---
    gemm_ln_kernel<DIM_H, 0><<<ggrid, 256, 0, stream>>>(
        dec_in, Wt3, bb[3], gg[3], be[3], act3, nullptr, DIM_H);
    gemm_ln_kernel<DIM_H, 0><<<ggrid, 256, 0, stream>>>(
        act3,  Wt4, bb[4], gg[4], be[4], act4, nullptr, DIM_H);
    gemm_ln_kernel<DIM_H, 2><<<dim3(BATCH / 32, DIM_IN / 1024), 256, 0, stream>>>(
        act4, Wt5, bb[5], nullptr, nullptr, nullptr, xhat, DIM_IN);

    finalize_loss_kernel<<<1, 1, 0, stream>>>(lossp, loss_out);
}
